// AutoRegressive_764504178982
// MI455X (gfx1250) — matrix operand order repaired and checked by reading, not compiled, not run on hardware
//
#include <hip/hip_runtime.h>
#include <cstdint>
#include <cstddef>

// ---------------- problem constants ----------------
#define BB 2
#define TT 256
#define TA 768
#define EE 225
#define CC 8
#define DD 1024
#define HH 16
#define FF 4096
#define LL 2
#define VV 256
#define KK 1024
#define SS (TT + TA + EE)   // 1249
#define DH 64               // D / H

// ---------------- CDNA5 WMMA types ----------------
typedef __attribute__((ext_vector_type(16))) __bf16          v16bf;
typedef __attribute__((ext_vector_type(8)))  float           v8f;
typedef __attribute__((ext_vector_type(8)))  unsigned short  v8us;

union Frag {
    v16bf v;
    v8us  h[2];
};

static __device__ __forceinline__ unsigned short f2bf(float f) {
    unsigned int u = __builtin_bit_cast(unsigned int, f);
    unsigned int r = u + 0x7FFFu + ((u >> 16) & 1u);   // round-to-nearest-even
    return (unsigned short)(r >> 16);
}

static __device__ __forceinline__ v8f wmma_bf16(v16bf a, v16bf b, v8f c) {
    // D = A(16x32 bf16) x B(32x16 bf16) + C(16x16 f32)
    return __builtin_amdgcn_wmma_f32_16x16x32_bf16(
        false, a, false, b, (short)0, c, false, false);
}

// ---------------- small helper kernels ----------------
__global__ __launch_bounds__(32) void sdyn_kernel(const int* __restrict__ tl,
                                                  const int* __restrict__ al,
                                                  int* __restrict__ out) {
    if (threadIdx.x == 0) {
        int m = 0;
        for (int b = 0; b < BB; ++b) {
            int v = tl[b] + al[b];
            m = v > m ? v : m;
        }
        *out = m;
    }
}

__global__ __launch_bounds__(256) void cvt_bf16_kernel(const float* __restrict__ src,
                                                       unsigned short* __restrict__ dst,
                                                       int n) {
    int i = blockIdx.x * 256 + threadIdx.x;
    if (i < n) dst[i] = f2bf(src[i]);
}

// pack embeddings: x[b,p,:] = text / audio / sum_c enrolled, else 0
__global__ __launch_bounds__(256) void pack_kernel(
    const int* __restrict__ text, const int* __restrict__ audio,
    const int* __restrict__ enr,
    const int* __restrict__ tl_, const int* __restrict__ al_,
    const float* __restrict__ text_emb, const float* __restrict__ audio_embs,
    float* __restrict__ x, unsigned short* __restrict__ xb,
    unsigned short* __restrict__ membf) {
    const int row = blockIdx.x;            // b*SS + p
    const int b = row / SS, p = row % SS;
    const int tl = tl_[b], al = al_[b];
    const int d0 = threadIdx.x * 4;

    float v[4] = {0.f, 0.f, 0.f, 0.f};
    if (p < tl) {
        const float* src = text_emb + (size_t)text[b * TT + p] * DD;
#pragma unroll
        for (int j = 0; j < 4; ++j) v[j] = src[d0 + j];
    } else if (p < tl + al) {
        const float* src = audio_embs + (size_t)audio[b * TA + (p - tl)] * DD;
#pragma unroll
        for (int j = 0; j < 4; ++j) v[j] = src[d0 + j];
    } else if (p < tl + al + EE) {
        const int e = p - tl - al;
        for (int c = 0; c < CC; ++c) {
            int idx = enr[((size_t)b * CC + c) * EE + e];
            const float* src = audio_embs + ((size_t)c * KK + idx) * DD;
#pragma unroll
            for (int j = 0; j < 4; ++j) v[j] += src[d0 + j];
        }
    }
    size_t o = (size_t)row * DD + d0;
#pragma unroll
    for (int j = 0; j < 4; ++j) {
        x[o + j] = v[j];
        unsigned short bv = f2bf(v[j]);
        xb[o + j] = bv;
        membf[o + j] = bv;
    }
}

// y = LN(x + delta)*g + b ; writes f32 back into x and bf16 copy into xb
__global__ __launch_bounds__(256) void residual_ln_kernel(
    float* __restrict__ x, const float* __restrict__ delta,
    const float* __restrict__ g, const float* __restrict__ bta,
    unsigned short* __restrict__ xb) {
    const int row = blockIdx.x;
    const int tid = threadIdx.x;
    const int lane = tid & 31, wid = tid >> 5;
    __shared__ float red[8];
    __shared__ float stat[2];

    float v[4];
    float* xr = x + (size_t)row * DD;
    const float* dr = delta + (size_t)row * DD;
#pragma unroll
    for (int j = 0; j < 4; ++j) v[j] = xr[tid * 4 + j] + dr[tid * 4 + j];

    float s = v[0] + v[1] + v[2] + v[3];
#pragma unroll
    for (int m = 16; m > 0; m >>= 1) s += __shfl_xor(s, m, 32);
    if (lane == 0) red[wid] = s;
    __syncthreads();
    if (tid == 0) {
        float t = 0.f;
        for (int i = 0; i < 8; ++i) t += red[i];
        stat[0] = t * (1.0f / DD);
    }
    __syncthreads();
    const float mean = stat[0];

    float q = 0.f;
#pragma unroll
    for (int j = 0; j < 4; ++j) { float d = v[j] - mean; q += d * d; }
#pragma unroll
    for (int m = 16; m > 0; m >>= 1) q += __shfl_xor(q, m, 32);
    __syncthreads();
    if (lane == 0) red[wid] = q;
    __syncthreads();
    if (tid == 0) {
        float t = 0.f;
        for (int i = 0; i < 8; ++i) t += red[i];
        stat[1] = rsqrtf(t * (1.0f / DD) + 1e-5f);
    }
    __syncthreads();
    const float inv = stat[1];
#pragma unroll
    for (int j = 0; j < 4; ++j) {
        int c = tid * 4 + j;
        float y = (v[j] - mean) * inv * g[c] + bta[c];
        xr[c] = y;
        xb[(size_t)row * DD + c] = f2bf(y);
    }
}

// ---------------- WMMA GEMM: out[M,N] = A[M,K] * W[N,K]^T + bias ----------------
// 128x128 block tile, BK=64, 8 waves in 2(M) x 4(N); each wave 64x32 ->
// 16 v_wmma per K-iteration. Global->VGPR register double buffering overlaps
// loads with the matrix pipe; LDS tiles feed the WMMA operand swizzles.
#define BM 128
#define BN 128
#define BK 64
#define LDK (BK + 8)   // LDS row stride in halfs (144B, 16B-aligned chunks)

__global__ __launch_bounds__(256) void gemm_kernel(
    const unsigned short* __restrict__ A, int lda,
    const unsigned short* __restrict__ W, int ldw,
    const float* __restrict__ bias,
    float* __restrict__ Cout, unsigned short* __restrict__ CoutBf, int ldc,
    int M, int N, int Kt, int relu) {
    __shared__ unsigned short As[BM][LDK];
    __shared__ unsigned short Ws[BN][LDK];

    const int tid = threadIdx.x;
    const int lane = tid & 31;
    const int wid = tid >> 5;      // 8 waves
    const int wm = wid & 1;        // 2 waves along M (64 rows each)
    const int wn = wid >> 1;       // 4 waves along N (32 cols each)
    const int lo = lane & 15, hi4 = lane >> 4;
    const int m0 = blockIdx.y * BM;
    const int n0 = blockIdx.x * BN;

    v8f acc[4][2];
#pragma unroll
    for (int i = 0; i < 4; ++i)
#pragma unroll
        for (int j = 0; j < 2; ++j)
#pragma unroll
            for (int e = 0; e < 8; ++e) acc[i][j][e] = 0.f;

    // staging map: 2 threads per row, 32 halfs (4 x v8us) each
    const int srow = tid >> 1;           // 0..127
    const int scol = (tid & 1) * 32;     // 0 or 32

    const int gmA = m0 + srow;
    const bool aval = gmA < M;
    const unsigned short* aPtr = A + (size_t)(aval ? gmA : 0) * lda + scol;
    const unsigned short* wPtr = W + (size_t)(n0 + srow) * ldw + scol;  // N % 128 == 0

    v8us aR[4], bR[4];
    {
        const v8us z = {0, 0, 0, 0, 0, 0, 0, 0};
#pragma unroll
        for (int c = 0; c < 4; ++c) {
            aR[c] = aval ? *(const v8us*)(aPtr + c * 8) : z;
            bR[c] = *(const v8us*)(wPtr + c * 8);
        }
    }

    for (int k0 = 0; k0 < Kt; k0 += BK) {
        // commit staged registers to LDS
#pragma unroll
        for (int c = 0; c < 4; ++c) {
            *(v8us*)&As[srow][scol + c * 8] = aR[c];
            *(v8us*)&Ws[srow][scol + c * 8] = bR[c];
        }
        __syncthreads();

        // prefetch next K-tile into registers while WMMAs run
        const bool more = (k0 + BK) < Kt;
        v8us aN[4], bN[4];
        if (more) {
            const v8us z = {0, 0, 0, 0, 0, 0, 0, 0};
            const unsigned short* ap = aPtr + (k0 + BK);
            const unsigned short* wp = wPtr + (k0 + BK);
            if (aval) __builtin_prefetch(ap + BK, 0, 1);
#pragma unroll
            for (int c = 0; c < 4; ++c) {
                aN[c] = aval ? *(const v8us*)(ap + c * 8) : z;
                bN[c] = *(const v8us*)(wp + c * 8);
            }
        }

        // 2 K-subtiles x (4 M x 2 N) = 16 WMMAs
#pragma unroll
        for (int ks = 0; ks < BK; ks += 32) {
            Frag af[4], wf[2];
#pragma unroll
            for (int sm = 0; sm < 4; ++sm) {
                int r = wm * 64 + sm * 16 + lo;
                int base = ks + hi4 * 8;              // A layout: K {hi*8..+7, +16..+23}
                af[sm].h[0] = *(const v8us*)&As[r][base];
                af[sm].h[1] = *(const v8us*)&As[r][base + 16];
            }
#pragma unroll
            for (int sn = 0; sn < 2; ++sn) {
                int c = wn * 32 + sn * 16 + lo;
                int base = ks + hi4 * 8;              // B layout: K {hi*8..+7, +16..+23}
                wf[sn].h[0] = *(const v8us*)&Ws[c][base];
                wf[sn].h[1] = *(const v8us*)&Ws[c][base + 16];
            }
#pragma unroll
            for (int sm = 0; sm < 4; ++sm)
#pragma unroll
                for (int sn = 0; sn < 2; ++sn)
                    acc[sm][sn] = wmma_bf16(af[sm].v, wf[sn].v, acc[sm][sn]);
        }
        __syncthreads();

        if (more) {
#pragma unroll
            for (int c = 0; c < 4; ++c) { aR[c] = aN[c]; bR[c] = bN[c]; }
        }
    }

    // epilogue: C layout — lane = N col (lo), rows = vgpr + 8*hi
#pragma unroll
    for (int sm = 0; sm < 4; ++sm)
#pragma unroll
        for (int sn = 0; sn < 2; ++sn)
#pragma unroll
            for (int v = 0; v < 8; ++v) {
                int ml = wm * 64 + sm * 16 + v + hi4 * 8;
                int nl = wn * 32 + sn * 16 + lo;
                int gm = m0 + ml, gn = n0 + nl;
                if (gm < M) {
                    float val = acc[sm][sn][v] + bias[gn];
                    if (relu) val = fmaxf(val, 0.f);
                    size_t idx = (size_t)gm * ldc + gn;
                    if (Cout) Cout[idx] = val;
                    if (CoutBf) CoutBf[idx] = f2bf(val);
                }
            }
}

// ---------------- flash attention (bf16 WMMA, streaming softmax) ----------------
// grid: (ceil(S/16), H, B), block: 32 (one wave). qkvb: [B,S,3D] bf16 (Q|K|V).
__global__ __launch_bounds__(32) void attn_kernel(
    const unsigned short* __restrict__ qkvb,
    unsigned short* __restrict__ obf,
    const int* __restrict__ sdyn_p,
    int causal) {
    __shared__ unsigned short Pt[16][40];    // softmax probs tile (16 q x 32 k)
    __shared__ unsigned short Vt[64][40];    // V^T tile: [dh][key]

    const int lane = threadIdx.x;
    const int lo = lane & 15, hi4 = lane >> 4;
    const int qt = blockIdx.x, h = blockIdx.y, b = blockIdx.z;
    const int q0 = qt * 16;
    const int sdyn = *sdyn_p;

    // Q fragments (A operand): 16 queries x 64 dh -> two K=32 fragments
    Frag qa[2];
    {
        int qrow = q0 + lo;
        if (qrow >= SS) qrow = SS - 1;
        const unsigned short* qptr = qkvb + ((size_t)(b * SS + qrow)) * (3 * DD) + h * DH;
#pragma unroll
        for (int j = 0; j < 2; ++j) {
            int base = j * 32 + hi4 * 8;
            qa[j].h[0] = *(const v8us*)(qptr + base);
            qa[j].h[1] = *(const v8us*)(qptr + base + 16);
        }
    }

    float mrun[8], lrun[8];
    v8f accO[4];
#pragma unroll
    for (int v = 0; v < 8; ++v) { mrun[v] = -1e30f; lrun[v] = 0.f; }
#pragma unroll
    for (int t = 0; t < 4; ++t)
#pragma unroll
        for (int v = 0; v < 8; ++v) accO[t][v] = 0.f;

    const int nkb = (SS + 31) / 32;
    for (int kb = 0; kb < nkb; ++kb) {
        const int k0 = kb * 32;

        // stage V^T: Vt[dh][key]
        {
            int key = k0 + lane;
            bool kv = key < SS;
            const unsigned short* vptr =
                qkvb + ((size_t)(b * SS + (kv ? key : SS - 1))) * (3 * DD) + 2 * DD + h * DH;
            v8us z = {0, 0, 0, 0, 0, 0, 0, 0};
#pragma unroll
            for (int d8 = 0; d8 < 8; ++d8) {
                v8us val = kv ? *(const v8us*)(vptr + d8 * 8) : z;
#pragma unroll
                for (int e = 0; e < 8; ++e) Vt[d8 * 8 + e][lane] = val[e];
            }
        }

        // scores s = Q K^T for two 16-key tiles
        v8f sc[2];
#pragma unroll
        for (int t = 0; t < 2; ++t) {
            int kn = k0 + t * 16 + lo;
            int krow = kn < SS ? kn : SS - 1;
            const unsigned short* kptr =
                qkvb + ((size_t)(b * SS + krow)) * (3 * DD) + DD + h * DH;
            v8f c = {0.f, 0.f, 0.f, 0.f, 0.f, 0.f, 0.f, 0.f};
#pragma unroll
            for (int j = 0; j < 2; ++j) {
                Frag kf;
                int base = j * 32 + hi4 * 8;
                kf.h[0] = *(const v8us*)(kptr + base);
                kf.h[1] = *(const v8us*)(kptr + base + 16);
                c = wmma_bf16(qa[j].v, kf.v, c);
            }
            sc[t] = c;
        }

        // online softmax (rows live across the 16 lanes of each half)
        float p[2][8];
#pragma unroll
        for (int v = 0; v < 8; ++v) {
            const int iq = q0 + v + hi4 * 8;
            float mx = -1e30f;
#pragma unroll
            for (int t = 0; t < 2; ++t) {
                int jk = k0 + t * 16 + lo;
                float val = sc[t][v] * 0.125f;               // 1/sqrt(64)
                if (causal && (jk > iq) && (jk >= sdyn)) val += -1e9f;
                if (jk >= SS) val = -1e30f;                  // padding mask
                p[t][v] = val;
                mx = fmaxf(mx, val);
            }
#pragma unroll
            for (int m = 1; m < 16; m <<= 1) mx = fmaxf(mx, __shfl_xor(mx, m, 32));
            float mnew = fmaxf(mrun[v], mx);
            float scale = __expf(mrun[v] - mnew);
            float rsum = 0.f;
#pragma unroll
            for (int t = 0; t < 2; ++t) {
                float e = __expf(p[t][v] - mnew);
                p[t][v] = e;
                rsum += e;
            }
#pragma unroll
            for (int m = 1; m < 16; m <<= 1) rsum += __shfl_xor(rsum, m, 32);
            lrun[v] = lrun[v] * scale + rsum;
            mrun[v] = mnew;
#pragma unroll
            for (int t2 = 0; t2 < 4; ++t2) accO[t2][v] *= scale;
        }

        // re-layout P into A-fragment form through LDS (wave-private, LDS in-order)
#pragma unroll
        for (int v = 0; v < 8; ++v)
#pragma unroll
            for (int t = 0; t < 2; ++t)
                Pt[v + hi4 * 8][t * 16 + lo] = f2bf(p[t][v]);

        Frag pf;
        {
            int base = hi4 * 8;
            pf.h[0] = *(const v8us*)&Pt[lo][base];
            pf.h[1] = *(const v8us*)&Pt[lo][base + 16];
        }

        // O += P * V over 4 dh tiles
#pragma unroll
        for (int t2 = 0; t2 < 4; ++t2) {
            Frag vf;
            int dh = t2 * 16 + lo;
            int base = hi4 * 8;
            vf.h[0] = *(const v8us*)&Vt[dh][base];
            vf.h[1] = *(const v8us*)&Vt[dh][base + 16];
            accO[t2] = wmma_bf16(pf.v, vf.v, accO[t2]);
        }
    }

    // epilogue: normalize and write bf16 attention output [B,S,D]
#pragma unroll
    for (int t2 = 0; t2 < 4; ++t2)
#pragma unroll
        for (int v = 0; v < 8; ++v) {
            int m = v + hi4 * 8;
            int q = q0 + m;
            if (q < SS) {
                float val = accO[t2][v] / lrun[v];
                obf[((size_t)(b * SS + q)) * DD + h * DH + t2 * 16 + lo] = f2bf(val);
            }
        }
}

// ---------------- host orchestration ----------------
extern "C" void kernel_launch(void* const* d_in, const int* in_sizes, int n_in,
                              void* d_out, int out_size, void* d_ws, size_t ws_size,
                              hipStream_t stream) {
    (void)in_sizes; (void)n_in; (void)out_size; (void)ws_size;

    const int*   text      = (const int*)d_in[0];
    const int*   audio     = (const int*)d_in[1];
    const int*   enr       = (const int*)d_in[2];
    const int*   tl        = (const int*)d_in[3];
    const int*   al        = (const int*)d_in[4];
    const float* text_emb  = (const float*)d_in[5];
    const float* audio_em  = (const float*)d_in[6];
    const float* sa_in_w   = (const float*)d_in[7];
    const float* sa_in_b   = (const float*)d_in[8];
    const float* sa_out_w  = (const float*)d_in[9];
    const float* sa_out_b  = (const float*)d_in[10];
    const float* ca_in_w   = (const float*)d_in[11];
    const float* ca_in_b   = (const float*)d_in[12];
    const float* ca_out_w  = (const float*)d_in[13];
    const float* ca_out_b  = (const float*)d_in[14];
    const float* ff_w1     = (const float*)d_in[15];
    const float* ff_b1     = (const float*)d_in[16];
    const float* ff_w2     = (const float*)d_in[17];
    const float* ff_b2     = (const float*)d_in[18];
    const float* ln1_g     = (const float*)d_in[19];
    const float* ln1_b     = (const float*)d_in[20];
    const float* ln2_g     = (const float*)d_in[21];
    const float* ln2_b     = (const float*)d_in[22];
    const float* ln3_g     = (const float*)d_in[23];
    const float* ln3_b     = (const float*)d_in[24];

    // workspace bump allocator
    char* ws = (char*)d_ws;
    size_t off = 0;
    auto alloc = [&](size_t bytes) -> char* {
        char* p = ws + off;
        off += (bytes + 255) & ~(size_t)255;
        return p;
    };

    const size_t BSD = (size_t)BB * SS * DD;   // 2,557,952
    const int M = BB * SS;                     // 2498

    float*          x     = (float*)alloc(BSD * 4);
    float*          tmp   = (float*)alloc(BSD * 4);
    unsigned short* qkvb  = (unsigned short*)alloc(BSD * 3 * 2);
    unsigned short* xb    = (unsigned short*)alloc(BSD * 2);
    unsigned short* membf = (unsigned short*)alloc(BSD * 2);
    unsigned short* obf   = (unsigned short*)alloc(BSD * 2);
    unsigned short* hbf   = (unsigned short*)alloc((size_t)BB * SS * FF * 2);
    unsigned short* w_sa_in  = (unsigned short*)alloc((size_t)LL * 3 * DD * DD * 2);
    unsigned short* w_sa_out = (unsigned short*)alloc((size_t)LL * DD * DD * 2);
    unsigned short* w_ca_in  = (unsigned short*)alloc((size_t)LL * 3 * DD * DD * 2);
    unsigned short* w_ca_out = (unsigned short*)alloc((size_t)LL * DD * DD * 2);
    unsigned short* w_ff1    = (unsigned short*)alloc((size_t)LL * FF * DD * 2);
    unsigned short* w_ff2    = (unsigned short*)alloc((size_t)LL * DD * FF * 2);
    int*            sdyn     = (int*)alloc(256);

    auto cvt = [&](const float* s, unsigned short* d, size_t n) {
        int blocks = (int)((n + 255) / 256);
        cvt_bf16_kernel<<<blocks, 256, 0, stream>>>(s, d, (int)n);
    };
    auto gemm = [&](const unsigned short* A, int lda, const unsigned short* W, int ldw,
                    const float* bias, float* C, unsigned short* Cb, int ldc,
                    int Mm, int N, int Kt, int relu) {
        dim3 g(N / BN, (Mm + BM - 1) / BM);
        gemm_kernel<<<g, 256, 0, stream>>>(A, lda, W, ldw, bias, C, Cb, ldc, Mm, N, Kt, relu);
    };

    sdyn_kernel<<<1, 32, 0, stream>>>(tl, al, sdyn);
    pack_kernel<<<BB * SS, 256, 0, stream>>>(text, audio, enr, tl, al, text_emb,
                                             audio_em, x, xb, membf);

    cvt(sa_in_w,  w_sa_in,  (size_t)LL * 3 * DD * DD);
    cvt(sa_out_w, w_sa_out, (size_t)LL * DD * DD);
    cvt(ca_in_w,  w_ca_in,  (size_t)LL * 3 * DD * DD);
    cvt(ca_out_w, w_ca_out, (size_t)LL * DD * DD);
    cvt(ff_w1,    w_ff1,    (size_t)LL * FF * DD);
    cvt(ff_w2,    w_ff2,    (size_t)LL * DD * FF);

    const dim3 ga((SS + 15) / 16, HH, BB);

    for (int l = 0; l < LL; ++l) {
        // ---- self attention ----
        gemm(xb, DD, w_sa_in + (size_t)l * 3 * DD * DD, DD, sa_in_b + (size_t)l * 3 * DD,
             nullptr, qkvb, 3 * DD, M, 3 * DD, DD, 0);
        attn_kernel<<<ga, 32, 0, stream>>>(qkvb, obf, sdyn, 1);
        gemm(obf, DD, w_sa_out + (size_t)l * DD * DD, DD, sa_out_b + (size_t)l * DD,
             tmp, nullptr, DD, M, DD, DD, 0);
        residual_ln_kernel<<<M, 256, 0, stream>>>(x, tmp, ln1_g + l * DD, ln1_b + l * DD, xb);

        // ---- cross attention (Q from x, K/V from packed memory) ----
        gemm(xb, DD, w_ca_in + (size_t)l * 3 * DD * DD, DD, ca_in_b + (size_t)l * 3 * DD,
             nullptr, qkvb, 3 * DD, M, DD, DD, 0);
        gemm(membf, DD, w_ca_in + (size_t)l * 3 * DD * DD + (size_t)DD * DD, DD,
             ca_in_b + (size_t)l * 3 * DD + DD,
             nullptr, qkvb + DD, 3 * DD, M, 2 * DD, DD, 0);
        attn_kernel<<<ga, 32, 0, stream>>>(qkvb, obf, sdyn, 0);
        gemm(obf, DD, w_ca_out + (size_t)l * DD * DD, DD, ca_out_b + (size_t)l * DD,
             tmp, nullptr, DD, M, DD, DD, 0);
        residual_ln_kernel<<<M, 256, 0, stream>>>(x, tmp, ln2_g + l * DD, ln2_b + l * DD, xb);

        // ---- FFN ----
        gemm(xb, DD, w_ff1 + (size_t)l * FF * DD, DD, ff_b1 + (size_t)l * FF,
             nullptr, hbf, FF, M, FF, DD, 1);
        gemm(hbf, FF, w_ff2 + (size_t)l * DD * FF, FF, ff_b2 + (size_t)l * DD,
             tmp, nullptr, DD, M, DD, FF, 0);
        residual_ln_kernel<<<M, 256, 0, stream>>>(x, tmp, ln3_g + l * DD, ln3_b + l * DD, xb);
    }

    hipMemcpyAsync(d_out, x, (size_t)M * DD * sizeof(float),
                   hipMemcpyDeviceToDevice, stream);
}
